// GFS_42614665511245
// MI455X (gfx1250) — matrix operand order repaired and checked by reading, not compiled, not run on hardware
//
#include <hip/hip_runtime.h>
#include <hip/hip_bf16.h>
#include <math.h>

#define NN   2048
#define TT   256
#define PDIM 5
#define TDIM 5
#define FF   128
#define EE   65536
#define GGB  5
#define G4   512   // 4*F gates

typedef __bf16 bf16;
typedef __attribute__((ext_vector_type(16))) __bf16 v16bf;
typedef __attribute__((ext_vector_type(8)))  float  v8f;

__device__ __forceinline__ float sigmoidf_(float x) { return 1.0f / (1.0f + expf(-x)); }

__device__ __forceinline__ v8f wmma_bf16(v16bf a, v16bf b, v8f c) {
  // D = A(16x32 bf16) * B(32x16 bf16) + C(16x16 f32)
  return __builtin_amdgcn_wmma_f32_16x16x32_bf16(false, a, false, b, (short)0, c, false, false);
}

// A fragment: 16x32 bf16 tile of row-major A[M, lda] at (row0, k0).
// Lane layout (ISA 7.12.2): lanes 0-15 rows, half h=lane>>4 selects K sub-blocks.
__device__ __forceinline__ v16bf load_a_frag(const bf16* A, int lda, int M, int row0, int k0) {
  int lane = threadIdx.x & 31;
  int r = lane & 15, h = lane >> 4;
  int row = row0 + r; if (row > M - 1) row = M - 1;   // clamp; stores are masked
  const bf16* p = A + (size_t)row * lda + k0;
  v16bf a;
#pragma unroll
  for (int i = 0; i < 8; ++i) a[i]     = p[h * 8 + i];
#pragma unroll
  for (int i = 0; i < 8; ++i) a[8 + i] = p[16 + h * 8 + i];
  return a;
}

// B fragment: 32x16 bf16 tile of row-major B[K, ldb] at (k0, col0).
// Lane l holds column n = col0 + (l&15); elements k = k0 + 8h + i and k0 + 16 + 8h + i.
__device__ __forceinline__ v16bf load_b_frag(const bf16* B, int ldb, int k0, int col0) {
  int lane = threadIdx.x & 31;
  int r = lane & 15, h = lane >> 4;
  const bf16* p = B + (size_t)k0 * ldb + col0 + r;
  v16bf b;
#pragma unroll
  for (int i = 0; i < 8; ++i) b[i]     = p[(size_t)(h * 8 + i) * ldb];
#pragma unroll
  for (int i = 0; i < 8; ++i) b[8 + i] = p[(size_t)(16 + h * 8 + i) * ldb];
  return b;
}

// ---------------------------------------------------------------------------
// Utility: f32 [R,C] -> bf16 [C,R] (transpose + cast), for WMMA B operands.
__global__ void transpose_cast_kernel(const float* __restrict__ W, bf16* __restrict__ Wt,
                                      int R, int C) {
  int idx = blockIdx.x * blockDim.x + threadIdx.x;
  if (idx >= R * C) return;
  int r = idx / C, c = idx % C;
  Wt[(size_t)c * R + r] = (bf16)W[idx];
}

// ---------------------------------------------------------------------------
// Fully fused LSTM step. Each wave owns a 16-row x 16-col slice of ALL FOUR
// gates (column tiles j0, 128+j0, 256+j0, 384+j0): 4 WMMA accumulator sets per
// K-step for the recurrent term (and for the dense layer-1 input term), then
// the cell update (sigmoid/tanh), c read-modify-write (exclusive per block),
// and the bf16 h write. h is ping-ponged (h_prev -> h_out) to avoid the
// cross-block RAW race on h rows.
__global__ void lstm_step_fused(const bf16* __restrict__ h_prev,       // [M,128] bf16
                                const bf16* __restrict__ WhhT,         // [128,512] bf16
                                const bf16* __restrict__ xt_bf,        // layer1 input (seq0 + t*F), or null
                                int lda_x,
                                const bf16* __restrict__ WihT,         // [128,512] bf16 (layer1)
                                const float* __restrict__ xt_f32,      // layer0 input (x + t*D), or null
                                int lda_xf, int D,
                                const float* __restrict__ Wih_f32,     // [512, D]
                                const float* __restrict__ bih,
                                const float* __restrict__ bhh,
                                float* __restrict__ c_state,           // [M,128]
                                bf16* __restrict__ h_out,              // [M,128] bf16
                                bf16* __restrict__ seq_t,              // seq + t*F (row stride T*F)
                                int M) {
  int j0   = blockIdx.x * 16;   // 0..112 within F
  int row0 = blockIdx.y * 16;
  v8f acc[4] = {};
#pragma unroll
  for (int k0 = 0; k0 < FF; k0 += 32) {
    v16bf a = load_a_frag(h_prev, FF, M, row0, k0);
#pragma unroll
    for (int g = 0; g < 4; ++g) {
      v16bf b = load_b_frag(WhhT, G4, k0, g * FF + j0);
      acc[g] = wmma_bf16(a, b, acc[g]);
    }
  }
  if (xt_bf) {  // uniform branch (SGPR arg): EXEC stays all-ones around WMMA
#pragma unroll
    for (int k0 = 0; k0 < FF; k0 += 32) {
      v16bf a = load_a_frag(xt_bf, lda_x, M, row0, k0);
#pragma unroll
      for (int g = 0; g < 4; ++g) {
        v16bf b = load_b_frag(WihT, G4, k0, g * FF + j0);
        acc[g] = wmma_bf16(a, b, acc[g]);
      }
    }
  }
  int lane = threadIdx.x & 31;
  int r = lane & 15, h = lane >> 4;
  int col = j0 + r;  // 0..127
  float b_i = bih[col]          + bhh[col];
  float b_f = bih[FF + col]     + bhh[FF + col];
  float b_g = bih[2 * FF + col] + bhh[2 * FF + col];
  float b_o = bih[3 * FF + col] + bhh[3 * FF + col];
#pragma unroll
  for (int i = 0; i < 8; ++i) {
    int row = row0 + h * 8 + i;
    if (row < M) {
      float gi = acc[0][i] + b_i;
      float gf = acc[1][i] + b_f;
      float gg = acc[2][i] + b_g;
      float go = acc[3][i] + b_o;
      if (xt_f32) {  // layer-0 tiny input dim: scalar MACs
        const float* xp = xt_f32 + (size_t)row * lda_xf;
        for (int k = 0; k < D; ++k) {
          float xv = xp[k];
          gi += xv * Wih_f32[(size_t)col * D + k];
          gf += xv * Wih_f32[(size_t)(FF + col) * D + k];
          gg += xv * Wih_f32[(size_t)(2 * FF + col) * D + k];
          go += xv * Wih_f32[(size_t)(3 * FF + col) * D + k];
        }
      }
      size_t ci = (size_t)row * FF + col;
      float cv = sigmoidf_(gf) * c_state[ci] + sigmoidf_(gi) * tanhf(gg);
      float hv = sigmoidf_(go) * tanhf(cv);
      c_state[ci] = cv;
      h_out[ci] = (bf16)hv;
      seq_t[(size_t)row * TT * FF + col] = (bf16)hv;
    }
  }
}

// ---------------------------------------------------------------------------
// Attention scores: s[n,t] = tanh(x[n,t,:] @ W1.T + b1) . W2 + b2
// One wave per (n, 16 timesteps): WMMA projection fused with tanh + W2 dot,
// cross-lane shuffle reduce over the 16 column-lanes.
__global__ void attn_scores(const bf16* __restrict__ seq,   // [M,T,128]
                            const bf16* __restrict__ W1T,   // [128,128] bf16
                            const float* __restrict__ b1,   // [128]
                            const float* __restrict__ W2,   // [128]
                            const float* __restrict__ b2,   // [1]
                            float* __restrict__ s,          // [M,T]
                            int M) {
  int n = blockIdx.x;
  int t0 = blockIdx.y * 16;
  const bf16* A = seq + (size_t)n * TT * FF;  // rows = t, lda = 128
  int lane = threadIdx.x & 31;
  int r = lane & 15, h = lane >> 4;
  float partial[8];
#pragma unroll
  for (int i = 0; i < 8; ++i) partial[i] = 0.0f;

  for (int c0 = 0; c0 < FF; c0 += 16) {
    v8f c = {};
#pragma unroll
    for (int k0 = 0; k0 < FF; k0 += 32) {
      v16bf a = load_a_frag(A, FF, TT, t0, k0);
      v16bf b = load_b_frag(W1T, FF, k0, c0);
      c = wmma_bf16(a, b, c);
    }
    int col = c0 + r;
    float w2 = W2[col], bb = b1[col];
#pragma unroll
    for (int i = 0; i < 8; ++i) partial[i] += tanhf(c[i] + bb) * w2;
  }
  // reduce across the 16 column lanes (xor<16 stays within each half)
#pragma unroll
  for (int off = 1; off < 16; off <<= 1)
#pragma unroll
    for (int i = 0; i < 8; ++i) partial[i] += __shfl_xor(partial[i], off, 32);
  if (r == 0) {
    float bb2 = b2[0];
#pragma unroll
    for (int i = 0; i < 8; ++i)
      s[(size_t)n * TT + t0 + h * 8 + i] = partial[i] + bb2;
  }
}

// Prefix-softmax context via closed form:
// out[n,d] = sum_tau e_tau * S_tau * x[n,tau,d], S_tau = sum_{t>=tau} 1/den_t.
__global__ void attn_context(const float* __restrict__ s, const bf16* __restrict__ seq,
                             float* __restrict__ out) {
  int n = blockIdx.x;
  int t = threadIdx.x;  // blockDim == TT == 256
  __shared__ float sh[TT];
  __shared__ float wts[TT];
  float sv = s[(size_t)n * TT + t];
  sh[t] = sv; __syncthreads();
  for (int o = 128; o > 0; o >>= 1) { if (t < o) sh[t] = fmaxf(sh[t], sh[t + o]); __syncthreads(); }
  float mx = sh[0]; __syncthreads();
  float e = expf(sv - mx);
  sh[t] = e; __syncthreads();
  for (int o = 1; o < TT; o <<= 1) {
    float v = (t >= o) ? sh[t - o] : 0.0f; __syncthreads();
    sh[t] += v; __syncthreads();
  }
  float den = sh[t]; __syncthreads();
  wts[t] = 1.0f / den; __syncthreads();
  sh[t] = wts[TT - 1 - t]; __syncthreads();          // reversed for suffix scan
  for (int o = 1; o < TT; o <<= 1) {
    float v = (t >= o) ? sh[t - o] : 0.0f; __syncthreads();
    sh[t] += v; __syncthreads();
  }
  float S = sh[TT - 1 - t]; __syncthreads();
  wts[t] = e * S; __syncthreads();
  if (t < FF) {
    float acc = 0.0f;
    const bf16* xp = seq + (size_t)n * TT * FF + t;
    for (int tau = 0; tau < TT; ++tau) acc += wts[tau] * (float)xp[(size_t)tau * FF];
    out[(size_t)n * FF + t] = acc;
  }
}

// ---------------------------------------------------------------------------
// GCN helpers (tiny GEMMs + edge aggregation: fp32 VALU, not worth WMMA)
__global__ void deg_init(float* __restrict__ deg, int n) {
  int i = blockIdx.x * blockDim.x + threadIdx.x;
  if (i < n) deg[i] = 1.0f;  // self-loop weight
}
__global__ void deg_edges(const int* __restrict__ ei, const float* __restrict__ w,
                          float* __restrict__ deg, int E) {
  int e = blockIdx.x * blockDim.x + threadIdx.x;
  if (e < E) atomicAdd(&deg[ei[E + e]], w[e]);   // col = ei[1][e]
}
__global__ void deg_finalize(float* __restrict__ deg, int n) {
  int i = blockIdx.x * blockDim.x + threadIdx.x;
  if (i < n) { float d = deg[i]; deg[i] = d > 0.0f ? rsqrtf(d) : 0.0f; }
}
__global__ void gemm_xWT(const float* __restrict__ X, const float* __restrict__ W,
                         float* __restrict__ Y, int M, int K, int Fo) {
  int idx = blockIdx.x * blockDim.x + threadIdx.x;
  if (idx >= M * Fo) return;
  int n = idx / Fo, j = idx % Fo;
  const float* x = X + (size_t)n * K;
  const float* w = W + (size_t)j * K;
  float acc = 0.0f;
  for (int k = 0; k < K; ++k) acc += x[k] * w[k];
  Y[idx] = acc;
}
__global__ void gcn_self(const float* __restrict__ h, const float* __restrict__ dinv,
                         const float* __restrict__ b, float* __restrict__ out, int M, int Fo) {
  int idx = blockIdx.x * blockDim.x + threadIdx.x;
  if (idx >= M * Fo) return;
  int c = idx / Fo, d = idx % Fo;
  out[idx] = dinv[c] * dinv[c] * h[idx] + b[d];
}
__global__ void gcn_edges(const int* __restrict__ ei, const float* __restrict__ w,
                          const float* __restrict__ dinv, const float* __restrict__ h,
                          float* __restrict__ out, int E, int Fo) {
  int idx = blockIdx.x * blockDim.x + threadIdx.x;
  if (idx >= E * Fo) return;
  int e = idx / Fo, d = idx % Fo;
  int row = ei[e], col = ei[E + e];
  float nw = dinv[row] * w[e] * dinv[col];
  atomicAdd(&out[(size_t)col * Fo + d], nw * h[(size_t)row * Fo + d]);
}

__global__ void final_mlp(const float* __restrict__ pg, const float* __restrict__ tm,
                          const float* __restrict__ Wm, const float* __restrict__ bm,
                          float* __restrict__ out, int M) {
  int n = blockIdx.x * blockDim.x + threadIdx.x;
  if (n >= M) return;
  float acc = bm[0];
  const float* p = pg + (size_t)n * FF;
  for (int d = 0; d < FF; ++d) acc += p[d] * Wm[d];
  const float* q = tm + (size_t)n * GGB;
  for (int j = 0; j < GGB; ++j) acc += q[j] * Wm[FF + j];
  out[n] = sigmoidf_(acc);
}

// ---------------------------------------------------------------------------
extern "C" void kernel_launch(void* const* d_in, const int* in_sizes, int n_in,
                              void* d_out, int out_size, void* d_ws, size_t ws_size,
                              hipStream_t stream) {
  (void)in_sizes; (void)n_in; (void)out_size; (void)ws_size;
  const float* price = (const float*)d_in[0];
  const float* gtr   = (const float*)d_in[1];
  const int*   p_ei  = (const int*)d_in[2];
  const float* p_w   = (const float*)d_in[3];
  const int*   t_ei  = (const int*)d_in[4];
  const float* t_w   = (const float*)d_in[5];
  const float* lp_Wih0 = (const float*)d_in[6];  const float* lp_Whh0 = (const float*)d_in[7];
  const float* lp_bih0 = (const float*)d_in[8];  const float* lp_bhh0 = (const float*)d_in[9];
  const float* lp_Wih1 = (const float*)d_in[10]; const float* lp_Whh1 = (const float*)d_in[11];
  const float* lp_bih1 = (const float*)d_in[12]; const float* lp_bhh1 = (const float*)d_in[13];
  const float* lt_Wih0 = (const float*)d_in[14]; const float* lt_Whh0 = (const float*)d_in[15];
  const float* lt_bih0 = (const float*)d_in[16]; const float* lt_bhh0 = (const float*)d_in[17];
  const float* lt_Wih1 = (const float*)d_in[18]; const float* lt_Whh1 = (const float*)d_in[19];
  const float* lt_bih1 = (const float*)d_in[20]; const float* lt_bhh1 = (const float*)d_in[21];
  const float* ap_W1 = (const float*)d_in[22]; const float* ap_b1 = (const float*)d_in[23];
  const float* ap_W2 = (const float*)d_in[24]; const float* ap_b2 = (const float*)d_in[25];
  const float* at_W1 = (const float*)d_in[26]; const float* at_b1 = (const float*)d_in[27];
  const float* at_W2 = (const float*)d_in[28]; const float* at_b2 = (const float*)d_in[29];
  const float* gp1_W = (const float*)d_in[30]; const float* gp1_b = (const float*)d_in[31];
  const float* gp2_W = (const float*)d_in[32]; const float* gp2_b = (const float*)d_in[33];
  const float* gt1_W = (const float*)d_in[34]; const float* gt1_b = (const float*)d_in[35];
  const float* gt2_W = (const float*)d_in[36]; const float* gt2_b = (const float*)d_in[37];
  const float* mlp_W = (const float*)d_in[38]; const float* mlp_b = (const float*)d_in[39];
  float* out = (float*)d_out;

  // ---- workspace carving -------------------------------------------------
  char* base = (char*)d_ws;
  size_t off = 0;
  auto carve = [&](size_t bytes) -> void* {
    void* p = base + off;
    off += (bytes + 255) & ~(size_t)255;
    return p;
  };
  bf16* seqA  = (bf16*)carve((size_t)NN * TT * FF * sizeof(bf16));   // price L0 out
  bf16* seqB  = (bf16*)carve((size_t)NN * TT * FF * sizeof(bf16));   // price L1 out
  bf16* seqTA = (bf16*)carve((size_t)GGB * TT * FF * sizeof(bf16));  // trend L0 out
  bf16* seqTB = (bf16*)carve((size_t)GGB * TT * FF * sizeof(bf16));  // trend L1 out
  bf16*  hbfA  = (bf16*)carve((size_t)NN * FF * sizeof(bf16));       // h ping
  bf16*  hbfB  = (bf16*)carve((size_t)NN * FF * sizeof(bf16));       // h pong
  float* cst   = (float*)carve((size_t)NN * FF * sizeof(float));
  float* scores = (float*)carve((size_t)NN * TT * sizeof(float));
  float* xatt  = (float*)carve((size_t)NN * FF * sizeof(float));
  float* gatt  = (float*)carve((size_t)GGB * FF * sizeof(float));
  bf16* lp_WhhT0 = (bf16*)carve((size_t)FF * G4 * sizeof(bf16));
  bf16* lp_WihT1 = (bf16*)carve((size_t)FF * G4 * sizeof(bf16));
  bf16* lp_WhhT1 = (bf16*)carve((size_t)FF * G4 * sizeof(bf16));
  bf16* lt_WhhT0 = (bf16*)carve((size_t)FF * G4 * sizeof(bf16));
  bf16* lt_WihT1 = (bf16*)carve((size_t)FF * G4 * sizeof(bf16));
  bf16* lt_WhhT1 = (bf16*)carve((size_t)FF * G4 * sizeof(bf16));
  bf16* apW1T = (bf16*)carve((size_t)FF * FF * sizeof(bf16));
  bf16* atW1T = (bf16*)carve((size_t)FF * FF * sizeof(bf16));
  float* degP = (float*)carve((size_t)NN * sizeof(float));
  float* degT = (float*)carve((size_t)NN * sizeof(float));
  float* h16  = (float*)carve((size_t)NN * 16 * sizeof(float));
  float* o16  = (float*)carve((size_t)NN * 16 * sizeof(float));
  float* h128 = (float*)carve((size_t)NN * FF * sizeof(float));
  float* pg   = (float*)carve((size_t)NN * FF * sizeof(float));
  float* tg   = (float*)carve((size_t)NN * FF * sizeof(float));
  float* trendM = (float*)carve((size_t)NN * GGB * sizeof(float));

  auto grid1 = [](int n) { return dim3((unsigned)((n + 255) / 256)); };

  // ---- weight transposes to bf16 (once) ----------------------------------
  transpose_cast_kernel<<<grid1(G4 * FF), 256, 0, stream>>>(lp_Whh0, lp_WhhT0, G4, FF);
  transpose_cast_kernel<<<grid1(G4 * FF), 256, 0, stream>>>(lp_Wih1, lp_WihT1, G4, FF);
  transpose_cast_kernel<<<grid1(G4 * FF), 256, 0, stream>>>(lp_Whh1, lp_WhhT1, G4, FF);
  transpose_cast_kernel<<<grid1(G4 * FF), 256, 0, stream>>>(lt_Whh0, lt_WhhT0, G4, FF);
  transpose_cast_kernel<<<grid1(G4 * FF), 256, 0, stream>>>(lt_Wih1, lt_WihT1, G4, FF);
  transpose_cast_kernel<<<grid1(G4 * FF), 256, 0, stream>>>(lt_Whh1, lt_WhhT1, G4, FF);
  transpose_cast_kernel<<<grid1(FF * FF), 256, 0, stream>>>(ap_W1, apW1T, FF, FF);
  transpose_cast_kernel<<<grid1(FF * FF), 256, 0, stream>>>(at_W1, atW1T, FF, FF);

  // ---- two-layer LSTM stack runner (fused step kernel, h ping-pong) ------
  auto run_layer = [&](const float* xf32, int lda_xf, int D, const float* Wih_f32,
                       const bf16* xseq, const bf16* WihT,
                       const bf16* WhhT, const float* bih, const float* bhh,
                       bf16* seq_out, int M) {
    dim3 ggrid(FF / 16, (unsigned)((M + 15) / 16));
    hipMemsetAsync(hbfA, 0, (size_t)M * FF * sizeof(bf16), stream);
    hipMemsetAsync(hbfB, 0, (size_t)M * FF * sizeof(bf16), stream);
    hipMemsetAsync(cst, 0, (size_t)M * FF * sizeof(float), stream);
    for (int t = 0; t < TT; ++t) {
      bf16* h_rd = (t & 1) ? hbfB : hbfA;
      bf16* h_wr = (t & 1) ? hbfA : hbfB;
      lstm_step_fused<<<ggrid, 32, 0, stream>>>(
          h_rd, WhhT,
          xseq ? (xseq + (size_t)t * FF) : nullptr, TT * FF, WihT,
          xf32 ? (xf32 + (size_t)t * D) : nullptr, lda_xf, D, Wih_f32,
          bih, bhh, cst, h_wr, seq_out + (size_t)t * FF, M);
    }
  };

  // price stack
  run_layer(price, TT * PDIM, PDIM, lp_Wih0, nullptr, nullptr,
            lp_WhhT0, lp_bih0, lp_bhh0, seqA, NN);
  run_layer(nullptr, 0, 0, nullptr, seqA, lp_WihT1,
            lp_WhhT1, lp_bih1, lp_bhh1, seqB, NN);
  // trend stack (M = GB = 5; clamped A-rows + masked stores)
  run_layer(gtr, TT * TDIM, TDIM, lt_Wih0, nullptr, nullptr,
            lt_WhhT0, lt_bih0, lt_bhh0, seqTA, GGB);
  run_layer(nullptr, 0, 0, nullptr, seqTA, lt_WihT1,
            lt_WhhT1, lt_bih1, lt_bhh1, seqTB, GGB);

  // ---- attention ----------------------------------------------------------
  attn_scores<<<dim3(NN, TT / 16), 32, 0, stream>>>(seqB, apW1T, ap_b1, ap_W2, ap_b2, scores, NN);
  attn_context<<<NN, TT, 0, stream>>>(scores, seqB, xatt);
  attn_scores<<<dim3(GGB, TT / 16), 32, 0, stream>>>(seqTB, atW1T, at_b1, at_W2, at_b2, scores, GGB);
  attn_context<<<GGB, TT, 0, stream>>>(scores, seqTB, gatt);

  // ---- GCN normalization (symmetric, with self-loops) ---------------------
  deg_init<<<grid1(NN), 256, 0, stream>>>(degP, NN);
  deg_edges<<<grid1(EE), 256, 0, stream>>>(p_ei, p_w, degP, EE);
  deg_finalize<<<grid1(NN), 256, 0, stream>>>(degP, NN);
  deg_init<<<grid1(NN), 256, 0, stream>>>(degT, NN);
  deg_edges<<<grid1(EE), 256, 0, stream>>>(t_ei, t_w, degT, EE);
  deg_finalize<<<grid1(NN), 256, 0, stream>>>(degT, NN);

  auto run_gcn = [&](const float* X, int K, const float* W, const float* b, int Fo,
                     const int* ei, const float* ew, const float* dinv,
                     float* H, float* OUT) {
    gemm_xWT<<<grid1(NN * Fo), 256, 0, stream>>>(X, W, H, NN, K, Fo);
    gcn_self<<<grid1(NN * Fo), 256, 0, stream>>>(H, dinv, b, OUT, NN, Fo);
    gcn_edges<<<grid1(EE * Fo), 256, 0, stream>>>(ei, ew, dinv, H, OUT, EE, Fo);
  };

  // price graph chain: x_att -> 16 -> 128
  run_gcn(xatt, FF, gp1_W, gp1_b, 16, p_ei, p_w, degP, h16, o16);
  run_gcn(o16, 16, gp2_W, gp2_b, FF, p_ei, p_w, degP, h128, pg);
  // trend graph chain (also fed with x_att, per reference)
  run_gcn(xatt, FF, gt1_W, gt1_b, 16, t_ei, t_w, degT, h16, o16);
  run_gcn(o16, 16, gt2_W, gt2_b, FF, t_ei, t_w, degT, h128, tg);

  // trend = tg @ g_att.T  ([N,128] x [5,128].T)
  gemm_xWT<<<grid1(NN * GGB), 256, 0, stream>>>(tg, gatt, trendM, NN, FF, GGB);

  // final MLP + sigmoid
  final_mlp<<<grid1(NN), 256, 0, stream>>>(pg, trendM, mlp_W, mlp_b, out, NN);
}
